// Encoder_14620068675922
// MI455X (gfx1250) — matrix operand order repaired and checked by reading, not compiled, not run on hardware
//
#include <hip/hip_runtime.h>
#include <hip/hip_bf16.h>

#define N_NODES 20000
#define E_EDGES 320000
#define EE      (E_EDGES + N_NODES)   // 340000 (with self loops)
#define HC      512
#define C_CH    128
#define H_HEADS 4
#define NEG     0.01f
#define ATT_NEG 0.2f

typedef __attribute__((ext_vector_type(16))) __bf16 v16bf;
typedef __attribute__((ext_vector_type(8)))  float  v8f;
typedef __attribute__((ext_vector_type(4)))  float  v4f;

// ---------------------------------------------------------------------------
// Fragment loaders (f32 memory -> bf16 WMMA operand registers).
// Operand layouts follow CDNA5 ISA 7.12.2 for V_WMMA_F32_16X16X32_BF16:
//   A (16x32): lane l (row = l&15): elems0-7 = K[h8..h8+7],
//              elems8-15 = K[16+h8..16+h8+7], h8 = (l>>4)*8
//   B (32x16): lane l (col = l&15): same K order as A (elems0-7 = K[h8..h8+7], elems8-15 = K[16+h8..16+h8+7])
//   C/D (16x16 f32): vgpr r, lane l -> row = r + (l>>4)*8, col = l&15
// ---------------------------------------------------------------------------
__device__ __forceinline__ v16bf load_frag_a(const float* __restrict__ p,
                                             int k0, int hl) {
    v4f x0 = *(const v4f*)(p + k0 + hl * 8);
    v4f x1 = *(const v4f*)(p + k0 + hl * 8 + 4);
    v4f x2 = *(const v4f*)(p + k0 + 16 + hl * 8);
    v4f x3 = *(const v4f*)(p + k0 + 16 + hl * 8 + 4);
    v16bf f;
#pragma unroll
    for (int i = 0; i < 4; ++i) {
        f[i]      = (__bf16)x0[i];
        f[4 + i]  = (__bf16)x1[i];
        f[8 + i]  = (__bf16)x2[i];
        f[12 + i] = (__bf16)x3[i];
    }
    return f;
}

__device__ __forceinline__ v16bf load_frag_b(const float* __restrict__ p,
                                             int k0, int hl) {
    v4f x0 = *(const v4f*)(p + k0 + hl * 8);
    v4f x1 = *(const v4f*)(p + k0 + hl * 8 + 4);
    v4f x2 = *(const v4f*)(p + k0 + 16 + hl * 8);
    v4f x3 = *(const v4f*)(p + k0 + 16 + hl * 8 + 4);
    v16bf f;
#pragma unroll
    for (int i = 0; i < 4; ++i) {
        f[i]      = (__bf16)x0[i];
        f[4 + i]  = (__bf16)x1[i];
        f[8 + i]  = (__bf16)x2[i];
        f[12 + i] = (__bf16)x3[i];
    }
    return f;
}

__device__ __forceinline__ void store_tile(float* __restrict__ Y, int M,
                                           int row_base, int col, int hl,
                                           v8f acc, float bias) {
#pragma unroll
    for (int r = 0; r < 8; ++r)
        Y[(size_t)(row_base + hl * 8 + r) * M + col] = acc[r] + bias;
}

// ---------------------------------------------------------------------------
// GEMM: Y[n,M] = X[n,K] @ W[M,K]^T + bias   (f32 in/out, bf16 WMMA compute)
// 2x2 register blocking: each wave owns a 32x32 macro-tile = 4 WMMA
// accumulators; each K-step loads 2 A + 2 B fragments and issues 4
// v_wmma_f32_16x16x32_bf16, reusing every fragment twice.
// Requires nrows % 32 == 0 and M % 32 == 0 (20000 and 512 both qualify).
// ---------------------------------------------------------------------------
__global__ __launch_bounds__(256) void gemm_bf16_wmma(
    float* __restrict__ Y, const float* __restrict__ X,
    const float* __restrict__ W, const float* __restrict__ Bias,
    int nrows, int K, int M)
{
    const int wave = threadIdx.x >> 5;
    const int lane = threadIdx.x & 31;
    const int tiles_n = M >> 5;                   // macro-tiles along N
    const int tile = blockIdx.x * 8 + wave;
    const int total = (nrows >> 5) * tiles_n;
    if (tile >= total) return;                    // wave-uniform (EXEC all-1 for WMMA)

    const int tm = tile / tiles_n;
    const int tn = tile - tm * tiles_n;
    const int r16 = lane & 15;
    const int hl  = lane >> 4;

    const int row0 = tm * 32 + r16;               // A rows for the two M tiles
    const int row1 = row0 + 16;
    const int col0 = tn * 32 + r16;               // B cols for the two N tiles
    const int col1 = col0 + 16;

    const float* __restrict__ a0p = X + (size_t)row0 * K;
    const float* __restrict__ a1p = X + (size_t)row1 * K;
    const float* __restrict__ b0p = W + (size_t)col0 * K;
    const float* __restrict__ b1p = W + (size_t)col1 * K;

    v8f acc00 = {}, acc01 = {}, acc10 = {}, acc11 = {};
    for (int k0 = 0; k0 < K; k0 += 32) {
        v16bf A0 = load_frag_a(a0p, k0, hl);
        v16bf A1 = load_frag_a(a1p, k0, hl);
        v16bf B0 = load_frag_b(b0p, k0, hl);
        v16bf B1 = load_frag_b(b1p, k0, hl);
        acc00 = __builtin_amdgcn_wmma_f32_16x16x32_bf16(false, A0, false, B0,
                                                        (short)0, acc00, false, false);
        acc01 = __builtin_amdgcn_wmma_f32_16x16x32_bf16(false, A0, false, B1,
                                                        (short)0, acc01, false, false);
        acc10 = __builtin_amdgcn_wmma_f32_16x16x32_bf16(false, A1, false, B0,
                                                        (short)0, acc10, false, false);
        acc11 = __builtin_amdgcn_wmma_f32_16x16x32_bf16(false, A1, false, B1,
                                                        (short)0, acc11, false, false);
    }

    const float bias0 = Bias ? Bias[col0] : 0.0f;
    const float bias1 = Bias ? Bias[col1] : 0.0f;
    store_tile(Y, M, tm * 32,      col0, hl, acc00, bias0);
    store_tile(Y, M, tm * 32,      col1, hl, acc01, bias1);
    store_tile(Y, M, tm * 32 + 16, col0, hl, acc10, bias0);
    store_tile(Y, M, tm * 32 + 16, col1, hl, acc11, bias1);
}

// ---------------------------------------------------------------------------
// Utility fills
// ---------------------------------------------------------------------------
__global__ void fill_f32(float* __restrict__ p, int n, float v) {
    int i = blockIdx.x * blockDim.x + threadIdx.x;
    if (i < n) p[i] = v;
}
__global__ void fill_u32(unsigned* __restrict__ p, int n, unsigned v) {
    int i = blockIdx.x * blockDim.x + threadIdx.x;
    if (i < n) p[i] = v;
}

// order-preserving float<->uint encode for atomicMax-based segment max
__device__ __forceinline__ unsigned enc_f32(float f) {
    unsigned b = __float_as_uint(f);
    return (b & 0x80000000u) ? ~b : (b | 0x80000000u);
}
__device__ __forceinline__ float dec_f32(unsigned u) {
    unsigned b = (u & 0x80000000u) ? (u & 0x7fffffffu) : ~u;
    return __uint_as_float(b);
}

__device__ __forceinline__ void edge_pair(const int* __restrict__ ei, int e,
                                          int& src, int& dst) {
    if (e < E_EDGES) { src = ei[e]; dst = ei[E_EDGES + e]; }
    else             { src = e - E_EDGES; dst = src; }       // self loops
}

// ---------------------------------------------------------------------------
// Edge kernel 1: logits[e][h] = sum_c att[h][c]*leaky(xl[src]+xr[dst], 0.2)
// 128 threads / edge: wave w -> head w, lane -> 4 consecutive channels.
// Also segment-max into m_enc via encoded atomicMax.
// ---------------------------------------------------------------------------
__global__ __launch_bounds__(256) void edge_logits_kernel(
    const float* __restrict__ xl, const float* __restrict__ xr,
    const int* __restrict__ ei, const float* __restrict__ att,
    float* __restrict__ logits, unsigned* __restrict__ m_enc)
{
    const int t = threadIdx.x & 127;
    const int e = blockIdx.x * 2 + (threadIdx.x >> 7);
    if (e >= EE) return;
    int src, dst; edge_pair(ei, e, src, dst);
    const int h = t >> 5, lane = t & 31;
    const int ch = h * C_CH + lane * 4;

    v4f lv = *(const v4f*)(xl + (size_t)src * HC + ch);
    v4f rv = *(const v4f*)(xr + (size_t)dst * HC + ch);
    v4f av = *(const v4f*)(att + ch);
    float s = 0.0f;
#pragma unroll
    for (int i = 0; i < 4; ++i) {
        float v = lv[i] + rv[i];
        v = (v > 0.0f) ? v : ATT_NEG * v;
        s += av[i] * v;
    }
#pragma unroll
    for (int off = 16; off > 0; off >>= 1) s += __shfl_xor(s, off, 32);
    if (lane == 0) {
        logits[(size_t)e * H_HEADS + h] = s;
        atomicMax(&m_enc[dst * H_HEADS + h], enc_f32(s));
    }
}

// ---------------------------------------------------------------------------
// Edge kernel 2: ez = exp(logit - m[dst]); denom[dst][h] += ez  (in-place)
// ---------------------------------------------------------------------------
__global__ void edge_ez_kernel(const int* __restrict__ ei,
                               float* __restrict__ ez_io,
                               const unsigned* __restrict__ m_enc,
                               float* __restrict__ denom)
{
    int idx = blockIdx.x * blockDim.x + threadIdx.x;
    if (idx >= EE * H_HEADS) return;
    const int e = idx >> 2, h = idx & 3;
    int src, dst; edge_pair(ei, e, src, dst); (void)src;
    const float m  = dec_f32(m_enc[dst * H_HEADS + h]);
    const float ez = __expf(ez_io[idx] - m);
    ez_io[idx] = ez;
    atomicAdd(&denom[dst * H_HEADS + h], ez);
}

// ---------------------------------------------------------------------------
// Edge kernel 3: alpha = ez/denom[dst]; write alpha out; agg[dst] += alpha*xl[src]
// ---------------------------------------------------------------------------
__global__ __launch_bounds__(256) void edge_scatter_kernel(
    const float* __restrict__ xl, const int* __restrict__ ei,
    const float* __restrict__ ez, const float* __restrict__ denom,
    float* __restrict__ alpha_out, float* __restrict__ agg)
{
    const int t = threadIdx.x & 127;
    const int e = blockIdx.x * 2 + (threadIdx.x >> 7);
    if (e >= EE) return;
    int src, dst; edge_pair(ei, e, src, dst);
    const int h = t >> 5, lane = t & 31;
    const float a = ez[(size_t)e * H_HEADS + h] / denom[dst * H_HEADS + h];
    if (lane == 0) alpha_out[(size_t)e * H_HEADS + h] = a;
    const int ch = h * C_CH + lane * 4;
    v4f xv = *(const v4f*)(xl + (size_t)src * HC + ch);
    float* dp = agg + (size_t)dst * HC + ch;
    atomicAdd(dp + 0, a * xv[0]);
    atomicAdd(dp + 1, a * xv[1]);
    atomicAdd(dp + 2, a * xv[2]);
    atomicAdd(dp + 3, a * xv[3]);
}

// ---------------------------------------------------------------------------
// BN column statistics: thread t privately owns columns t, t+256; coalesced
// row sweep over a block-sized row range; one atomicAdd per owned column.
// ---------------------------------------------------------------------------
__global__ __launch_bounds__(256) void col_stats_kernel(
    const float* __restrict__ X, float* __restrict__ sums,
    float* __restrict__ sumsq, int nrows, int ncols, int rows_per_block)
{
    const int c0 = threadIdx.x;
    const int r0 = blockIdx.x * rows_per_block;
    int r1 = r0 + rows_per_block; if (r1 > nrows) r1 = nrows;
    float s[2] = {0.f, 0.f}, q[2] = {0.f, 0.f};
    for (int r = r0; r < r1; ++r) {
        const float* __restrict__ row = X + (size_t)r * ncols;
        for (int j = 0, c = c0; c < ncols; c += 256, ++j) {
            float v = row[c];
            s[j] += v; q[j] += v * v;
        }
    }
    for (int j = 0, c = c0; c < ncols; c += 256, ++j) {
        atomicAdd(&sums[c],  s[j]);
        atomicAdd(&sumsq[c], q[j]);
    }
}

// ---------------------------------------------------------------------------
// BN normalize + leaky_relu(0.01):  y = lrelu(g*(x-mu)*rsqrt(var+eps)+b)
// ---------------------------------------------------------------------------
__global__ void bn_norm_kernel(const float* __restrict__ X, float* __restrict__ Y,
                               const float* __restrict__ sums,
                               const float* __restrict__ sumsq,
                               const float* __restrict__ g,
                               const float* __restrict__ b,
                               int nrows, int ncols)
{
    int idx = blockIdx.x * blockDim.x + threadIdx.x;
    if (idx >= nrows * ncols) return;
    const int c = idx % ncols;
    const float inv_n = 1.0f / (float)nrows;
    const float mu  = sums[c] * inv_n;
    const float var = sumsq[c] * inv_n - mu * mu;
    float v = g[c] * (X[idx] - mu) * rsqrtf(var + 1e-5f) + b[c];
    Y[idx] = (v > 0.0f) ? v : NEG * v;
}

// Layer 3: mean over heads -> (N, 128)
__global__ void head_mean_kernel(const float* __restrict__ agg,
                                 float* __restrict__ zpre)
{
    int idx = blockIdx.x * blockDim.x + threadIdx.x;
    if (idx >= N_NODES * C_CH) return;
    const int n = idx / C_CH, c = idx - n * C_CH;
    const float* a = agg + (size_t)n * HC + c;
    zpre[idx] = 0.25f * (a[0] + a[C_CH] + a[2 * C_CH] + a[3 * C_CH]);
}

// ---------------------------------------------------------------------------
extern "C" void kernel_launch(void* const* d_in, const int* in_sizes, int n_in,
                              void* d_out, int out_size, void* d_ws, size_t ws_size,
                              hipStream_t stream)
{
    (void)in_sizes; (void)n_in; (void)out_size; (void)ws_size;
    const float* x     = (const float*)d_in[0];
    const int*   ei    = (const int*)  d_in[1];
    const float* w1_l  = (const float*)d_in[3];
    const float* b1_l  = (const float*)d_in[4];
    const float* w1_r  = (const float*)d_in[5];
    const float* b1_r  = (const float*)d_in[6];
    const float* att1  = (const float*)d_in[7];
    const float* bn1_g = (const float*)d_in[9];
    const float* bn1_b = (const float*)d_in[10];
    const float* w2_l  = (const float*)d_in[11];
    const float* b2_l  = (const float*)d_in[12];
    const float* w2_r  = (const float*)d_in[13];
    const float* b2_r  = (const float*)d_in[14];
    const float* att2  = (const float*)d_in[15];
    const float* bn2_g = (const float*)d_in[17];
    const float* bn2_b = (const float*)d_in[18];
    const float* w3_l  = (const float*)d_in[19];
    const float* b3_l  = (const float*)d_in[20];
    const float* w3_r  = (const float*)d_in[21];
    const float* b3_r  = (const float*)d_in[22];
    const float* att3  = (const float*)d_in[23];
    const float* bn3_g = (const float*)d_in[25];
    const float* bn3_b = (const float*)d_in[26];
    // spatial/MHA/gate weights (d_in[27..36]) are dead code: softmax over a
    // single logit is identically 1, so z_final == z and sp never matters.

    // workspace layout (floats)
    float*    B_xl   = (float*)d_ws;                     // N*HC
    float*    B_xr   = B_xl + (size_t)N_NODES * HC;      // N*HC
    float*    B_h    = B_xr + (size_t)N_NODES * HC;      // N*HC (h / agg ping)
    float*    ezb    = B_h + (size_t)N_NODES * HC;       // EE*H
    unsigned* m_enc  = (unsigned*)(ezb + (size_t)EE * H_HEADS); // N*H
    float*    denom  = (float*)(m_enc + N_NODES * H_HEADS);     // N*H
    float*    colsum = denom + N_NODES * H_HEADS;        // 512
    float*    colsq  = colsum + 512;                     // 512
    float*    zpre   = colsq + 512;                      // N*C

    float* out    = (float*)d_out;
    float* out_z  = out;                                         // N*C
    float* out_a1 = out + (size_t)N_NODES * C_CH;                // EE*H
    float* out_a2 = out_a1 + (size_t)EE * H_HEADS;
    float* out_a3 = out_a2 + (size_t)EE * H_HEADS;
    float* out_g  = out_a3 + (size_t)EE * H_HEADS;               // N*1

    const int gemm_tiles  = (N_NODES / 32) * (HC / 32);  // 32x32 macro-tiles
    const int gemm_blocks = (gemm_tiles + 7) / 8;
    const int edge_blocks = (EE + 1) / 2;
    const int NEGINF_ENC  = 0x007FFFFF;  // enc(-inf)
    const int rpb = 100;                 // rows per block for stats (200 blocks)

    auto layer = [&](const float* Hin, const float* wl, const float* bl,
                     const float* wr, const float* br, const float* att,
                     float* alpha_out) {
        gemm_bf16_wmma<<<gemm_blocks, 256, 0, stream>>>(B_xl, Hin, wl, bl, N_NODES, HC, HC);
        gemm_bf16_wmma<<<gemm_blocks, 256, 0, stream>>>(B_xr, Hin, wr, br, N_NODES, HC, HC);
        fill_f32<<<((size_t)N_NODES * HC + 255) / 256, 256, 0, stream>>>(B_h, N_NODES * HC, 0.0f);
        fill_u32<<<(N_NODES * H_HEADS + 255) / 256, 256, 0, stream>>>(m_enc, N_NODES * H_HEADS, NEGINF_ENC);
        fill_f32<<<(N_NODES * H_HEADS + 255) / 256, 256, 0, stream>>>(denom, N_NODES * H_HEADS, 0.0f);
        edge_logits_kernel<<<edge_blocks, 256, 0, stream>>>(B_xl, B_xr, ei, att, ezb, m_enc);
        edge_ez_kernel<<<(EE * H_HEADS + 255) / 256, 256, 0, stream>>>(ei, ezb, m_enc, denom);
        edge_scatter_kernel<<<edge_blocks, 256, 0, stream>>>(B_xl, ei, ezb, denom, alpha_out, B_h);
    };

    auto bn = [&](const float* X, float* Y, const float* g, const float* b, int ncols) {
        fill_f32<<<2, 256, 0, stream>>>(colsum, 512, 0.0f);
        fill_f32<<<2, 256, 0, stream>>>(colsq, 512, 0.0f);
        col_stats_kernel<<<(N_NODES + rpb - 1) / rpb, 256, 0, stream>>>(X, colsum, colsq, N_NODES, ncols, rpb);
        bn_norm_kernel<<<((size_t)N_NODES * ncols + 255) / 256, 256, 0, stream>>>(
            X, Y, colsum, colsq, g, b, N_NODES, ncols);
    };

    // ---- layer 1 (K = 512 input features) ----
    layer(x, w1_l, b1_l, w1_r, b1_r, att1, out_a1);
    bn(B_h, B_h, bn1_g, bn1_b, HC);        // B_h becomes h1

    // ---- layer 2 ----
    layer(B_h, w2_l, b2_l, w2_r, b2_r, att2, out_a2);
    bn(B_h, B_h, bn2_g, bn2_b, HC);        // B_h becomes h2

    // ---- layer 3 (concat=False -> head mean) ----
    layer(B_h, w3_l, b3_l, w3_r, b3_r, att3, out_a3);
    head_mean_kernel<<<(N_NODES * C_CH + 255) / 256, 256, 0, stream>>>(B_h, zpre);
    bn(zpre, out_z, bn3_g, bn3_b, C_CH);   // z straight into d_out

    // gate = softmax over a length-1 axis == 1.0 everywhere
    fill_f32<<<(N_NODES + 255) / 256, 256, 0, stream>>>(out_g, N_NODES, 1.0f);
}
